// DiscriminativeLoss_21878563406298
// MI455X (gfx1250) — matrix operand order repaired and checked by reading, not compiled, not run on hardware
//
#include <hip/hip_runtime.h>
#include <hip/hip_bf16.h>
#include <math.h>

// ---------------------------------------------------------------------------
// DiscriminativeLoss on MI455X (gfx1250, wave32).
// Pass 1: segment sums+counts as one-hot matmul via V_WMMA_F32_16X16X32_BF16
//         (32 pixels per wmma; one-hot + count row exact in bf16, f32 accum).
// Pass 2: per-pixel distance-to-mean + reg norm (streaming, LDS bins).
// Tiny kernels: zero ws, compute means, finalize per-image + global mean.
// ---------------------------------------------------------------------------

#define NIMG      8
#define NCH       4
#define NSEG      17          // 0 = background, 1..16 = instances
#define DELTA_VAR 0.5f
#define DELTA_DIST 1.5f
#define GAMMA_REG 0.1f

// workspace layout (floats, per image)
#define WS_ACC    0           // 5 x 16 : rows 0..3 = channel sums, row 4 = counts
#define WS_MEANS  80          // 17 x 4 (seg 0 = zeros)
#define WS_DSUM   148         // 17 : per-seg sum of distances
#define WS_REG    165         // 1  : sum of ||e|| over valid pixels
#define WS_STRIDE 168

typedef float          v8f   __attribute__((ext_vector_type(8)));
typedef __bf16         v16bf __attribute__((ext_vector_type(16)));
typedef unsigned int   v8u   __attribute__((ext_vector_type(8)));
typedef short          i16x2 __attribute__((ext_vector_type(2)));
typedef unsigned short u16x2 __attribute__((ext_vector_type(2)));

__global__ void dl_zero_ws(float* ws, int n) {
    int i = blockIdx.x * blockDim.x + threadIdx.x;
    if (i < n) ws[i] = 0.0f;
}

// truncate-pack two f32 into a bf16 pair (lo -> [15:0], hi -> [31:16])
static __device__ __forceinline__ unsigned int pkbf(float lo, float hi) {
    unsigned int a = __builtin_bit_cast(unsigned int, lo);
    unsigned int b = __builtin_bit_cast(unsigned int, hi);
    return (a >> 16) | (b & 0xFFFF0000u);
}

// ---------------------------------------------------------------------------
// Pass 1: D[value, seg] += A(16x32 bf16 data) x B(32x16 bf16 onehot)
// 32 pixels per wmma. blockDim.x = 256 (8 waves).
// ---------------------------------------------------------------------------
__global__ void dl_pass1_wmma(const float* __restrict__ emb,
                              const int*   __restrict__ mask,
                              float* __restrict__ ws,
                              int blocksPerImage, int pxPerBlock, int hw) {
    const int img  = blockIdx.x / blocksPerImage;
    const int blk  = blockIdx.x % blocksPerImage;
    const int lane = threadIdx.x & 31;
    const int wave = __builtin_amdgcn_readfirstlane((int)(threadIdx.x >> 5));
    const int wavesPerBlock = blockDim.x >> 5;
    const int pxPerWave = pxPerBlock / wavesPerBlock;
    const int p0 = blk * pxPerBlock + wave * pxPerWave;

    const float* eimg = emb + (size_t)img * NCH * hw;
    const int*   mimg = mask + (size_t)img * hw;

    const bool hi   = lane >= 16;
    const int  segn = (lane & 15) + 1;         // B column -> segment id 1..16
    const int  mrow = lane & 15;               // A value-row this lane carries
    const int  crow = (mrow < NCH) ? mrow : (NCH - 1);
    // lo lanes cover K = 0..15 ; hi lanes K = 16..31
    const float* abase = eimg + (size_t)crow * hw + (hi ? 16 : 0);

    const i16x2 seg2   = { (short)segn, (short)segn };
    const u16x2 one2   = { 1, 1 };
    const u16x2 negc2  = { (unsigned short)0xC080u, (unsigned short)0xC080u };
    const u16x2 c1v    = { (unsigned short)0x3F80u, (unsigned short)0x3F80u };

    // A registers: loop-invariant constants for rows >= 4 (count row = 1.0bf16)
    const unsigned int initA = (mrow == NCH) ? 0x3F803F80u : 0u;
    v8u apk = { initA, initA, initA, initA, initA, initA, initA, initA };

    v8f c = {};
    for (int it = 0; it < pxPerWave; it += 32) {
        const int p = p0 + it;

        // ---- 32 wave-uniform mask values (scalar loads) -> 16-bit pairs ----
        const int4* mp = (const int4*)(mimg + p);
        int4 q0 = mp[0], q1 = mp[1], q2 = mp[2], q3 = mp[3];
        int4 q4 = mp[4], q5 = mp[5], q6 = mp[6], q7 = mp[7];
        int lopk[8], hipk[8];
        lopk[0] = q0.x | (q0.y << 16); lopk[1] = q0.z | (q0.w << 16);
        lopk[2] = q1.x | (q1.y << 16); lopk[3] = q1.z | (q1.w << 16);
        lopk[4] = q2.x | (q2.y << 16); lopk[5] = q2.z | (q2.w << 16);
        lopk[6] = q3.x | (q3.y << 16); lopk[7] = q3.z | (q3.w << 16);
        hipk[0] = q4.x | (q4.y << 16); hipk[1] = q4.z | (q4.w << 16);
        hipk[2] = q5.x | (q5.y << 16); hipk[3] = q5.z | (q5.w << 16);
        hipk[4] = q6.x | (q6.y << 16); hipk[5] = q6.z | (q6.w << 16);
        hipk[6] = q7.x | (q7.y << 16); hipk[7] = q7.z | (q7.w << 16);

        // ---- B = onehot(mask == seg) as packed bf16 {0,1} pairs ----
        v8u bpk;
#pragma unroll
        for (int j = 0; j < 8; ++j) {
            const int ps = hi ? hipk[j] : lopk[j];           // this lane's 2 masks
            i16x2 d  = __builtin_bit_cast(i16x2, ps) - seg2; // 0 iff match
            u16x2 ud = __builtin_bit_cast(u16x2, d);
            u16x2 u  = __builtin_elementwise_min(ud, one2);  // 0 if match else 1
            u16x2 r  = u * negc2 + c1v;                      // match->0x3F80 else 0
            bpk[j] = __builtin_bit_cast(unsigned int, r);
        }

        // ---- A = data rows (ch 0..3); rows 4..15 keep their constants ----
        if (mrow < NCH) {
            const float* ab = abase + p;
            float4 f0 = *(const float4*)(ab + 0);
            float4 f1 = *(const float4*)(ab + 4);
            float4 f2 = *(const float4*)(ab + 8);
            float4 f3 = *(const float4*)(ab + 12);
            apk[0] = pkbf(f0.x, f0.y); apk[1] = pkbf(f0.z, f0.w);
            apk[2] = pkbf(f1.x, f1.y); apk[3] = pkbf(f1.z, f1.w);
            apk[4] = pkbf(f2.x, f2.y); apk[5] = pkbf(f2.z, f2.w);
            apk[6] = pkbf(f3.x, f3.y); apk[7] = pkbf(f3.z, f3.w);
        }

        const v16bf A = __builtin_bit_cast(v16bf, apk);
        const v16bf B = __builtin_bit_cast(v16bf, bpk);
        c = __builtin_amdgcn_wmma_f32_16x16x32_bf16(false, A, false, B,
                                                    (short)0, c, false, false);
    }

    // D row m lives in VGPR m, lanes 0..15 (N = lane = seg-1). Flush rows 0..4.
    if (lane < 16) {
        float* acc = ws + (size_t)img * WS_STRIDE + WS_ACC;
#pragma unroll
        for (int m = 0; m < 5; ++m) {
            __hip_atomic_fetch_add(&acc[m * 16 + lane], c[m],
                                   __ATOMIC_RELAXED, __HIP_MEMORY_SCOPE_AGENT);
        }
    }
}

// ---------------------------------------------------------------------------
// Means: means[s][c] = present ? sums/counts : 0  (17x4 per image, seg 0 = 0)
// ---------------------------------------------------------------------------
__global__ void dl_means(float* ws) {
    const int img = blockIdx.x;
    const int t   = threadIdx.x;
    if (t >= NSEG * NCH) return;
    const int s = t >> 2, ch = t & 3;
    float* w = ws + (size_t)img * WS_STRIDE;
    float v = 0.0f;
    if (s > 0) {
        const float cnt = w[WS_ACC + 4 * 16 + (s - 1)];
        const float sum = w[WS_ACC + ch * 16 + (s - 1)];
        v = (cnt > 0.0f) ? (sum / cnt) : 0.0f;
    }
    w[WS_MEANS + t] = v;
}

// ---------------------------------------------------------------------------
// Pass 2: per-pixel dist-to-mean and reg norm. blockDim.x = 256.
// ---------------------------------------------------------------------------
__global__ void dl_pass2(const float* __restrict__ emb,
                         const int*   __restrict__ mask,
                         float* __restrict__ ws,
                         int blocksPerImage, int pxPerBlock, int hw) {
    __shared__ float lmeans[NSEG * NCH];
    __shared__ float bins[NSEG + 1];      // [0..16] dist sums, [17] reg sum

    const int img = blockIdx.x / blocksPerImage;
    const int blk = blockIdx.x % blocksPerImage;
    const int t   = threadIdx.x;
    float* w = ws + (size_t)img * WS_STRIDE;

    if (t < NSEG * NCH) lmeans[t] = w[WS_MEANS + t];
    if (t < NSEG + 1)   bins[t] = 0.0f;
    __syncthreads();

    const float* eimg = emb + (size_t)img * NCH * hw;
    const int*   mimg = mask + (size_t)img * hw;
    const int base = blk * pxPerBlock;

    float reg = 0.0f;
    for (int p = base + t; p < base + pxPerBlock; p += blockDim.x) {
        const int m = mimg[p];
        const float e0 = eimg[p];
        const float e1 = eimg[p + hw];
        const float e2 = eimg[p + 2 * hw];
        const float e3 = eimg[p + 3 * hw];
        if (m != 0) {
            const float d0 = e0 - lmeans[m * 4 + 0];
            const float d1 = e1 - lmeans[m * 4 + 1];
            const float d2 = e2 - lmeans[m * 4 + 2];
            const float d3 = e3 - lmeans[m * 4 + 3];
            const float dd = d0 * d0 + d1 * d1 + d2 * d2 + d3 * d3;
            const float dist = (dd > 0.0f) ? __builtin_sqrtf(dd) : 0.0f;
            __hip_atomic_fetch_add(&bins[m], dist,
                                   __ATOMIC_RELAXED, __HIP_MEMORY_SCOPE_WORKGROUP);
            const float nn = e0 * e0 + e1 * e1 + e2 * e2 + e3 * e3;
            reg += (nn > 0.0f) ? __builtin_sqrtf(nn) : 0.0f;
        }
    }

    // wave-level reduce reg (wave32), then one LDS add per wave
    for (int off = 16; off > 0; off >>= 1) reg += __shfl_down(reg, off);
    if ((t & 31) == 0)
        __hip_atomic_fetch_add(&bins[NSEG], reg,
                               __ATOMIC_RELAXED, __HIP_MEMORY_SCOPE_WORKGROUP);
    __syncthreads();

    if (t < NSEG)
        __hip_atomic_fetch_add(&w[WS_DSUM + t], bins[t],
                               __ATOMIC_RELAXED, __HIP_MEMORY_SCOPE_AGENT);
    if (t == NSEG)
        __hip_atomic_fetch_add(&w[WS_REG], bins[NSEG],
                               __ATOMIC_RELAXED, __HIP_MEMORY_SCOPE_AGENT);
}

// ---------------------------------------------------------------------------
// Finalize: per-image loss (var + pairwise-dist + reg), then mean over images.
// ---------------------------------------------------------------------------
__global__ void dl_finalize(const float* __restrict__ ws, float* __restrict__ out) {
    __shared__ float sloss[NIMG];
    __shared__ float shas[NIMG];
    const int t = threadIdx.x;
    if (t < NIMG) {
        const float* w = ws + (size_t)t * WS_STRIDE;
        float counts[NSEG];
        bool  present[NSEG];
        counts[0] = 0.0f; present[0] = false;
        float n = 0.0f, nvalid = 0.0f;
        for (int s = 1; s < NSEG; ++s) {
            counts[s]  = w[WS_ACC + 4 * 16 + (s - 1)];
            present[s] = counts[s] > 0.0f;
            if (present[s]) n += 1.0f;
            nvalid += counts[s];
        }
        // variance term
        float var = 0.0f;
        for (int s = 1; s < NSEG; ++s) {
            if (present[s]) {
                const float md = w[WS_DSUM + s] / fmaxf(counts[s], 1.0f);
                float r = md - DELTA_VAR;
                r = (r > 0.0f) ? r : 0.0f;
                var += r * r;
            }
        }
        var /= fmaxf(n, 1.0f);
        // pairwise distance term (diagonal contributes via +1e6 inside relu,
        // faithful to the reference)
        float dl = 0.0f;
        if (n > 1.0f) {
            float tot = 0.0f;
            for (int i = 1; i < NSEG; ++i) {
                if (!present[i]) continue;
                const float mi0 = w[WS_MEANS + i * 4 + 0];
                const float mi1 = w[WS_MEANS + i * 4 + 1];
                const float mi2 = w[WS_MEANS + i * 4 + 2];
                const float mi3 = w[WS_MEANS + i * 4 + 3];
                for (int j = 1; j < NSEG; ++j) {
                    if (!present[j]) continue;
                    const float q0 = mi0 - w[WS_MEANS + j * 4 + 0];
                    const float q1 = mi1 - w[WS_MEANS + j * 4 + 1];
                    const float q2 = mi2 - w[WS_MEANS + j * 4 + 2];
                    const float q3 = mi3 - w[WS_MEANS + j * 4 + 3];
                    const float d2 = q0 * q0 + q1 * q1 + q2 * q2 + q3 * q3;
                    const float pd = (d2 > 0.0f) ? __builtin_sqrtf(d2) : 0.0f;
                    float term = DELTA_DIST - pd + ((i == j) ? 1.0e6f : 0.0f);
                    term = (term > 0.0f) ? term : 0.0f;
                    tot += term * term;
                }
            }
            dl = tot / fmaxf(n * (n - 1.0f), 1.0f);
        }
        const float rl = w[WS_REG] / fmaxf(nvalid, 1.0f);
        sloss[t] = var + dl + GAMMA_REG * rl;
        shas[t]  = (nvalid > 0.0f) ? 1.0f : 0.0f;
    }
    __syncthreads();
    if (t == 0) {
        float s = 0.0f, h = 0.0f;
        for (int i = 0; i < NIMG; ++i) { s += sloss[i] * shas[i]; h += shas[i]; }
        out[0] = (h > 0.0f) ? (s / fmaxf(h, 1.0f)) : 0.0f;
    }
}

// ---------------------------------------------------------------------------
extern "C" void kernel_launch(void* const* d_in, const int* in_sizes, int n_in,
                              void* d_out, int out_size, void* d_ws, size_t ws_size,
                              hipStream_t stream) {
    const float* emb  = (const float*)d_in[0];     // (8, 4, 1024, 1024) f32
    const int*   mask = (const int*)d_in[1];       // (8, 1024, 1024) int
    float* out = (float*)d_out;
    float* ws  = (float*)d_ws;

    const int hw = in_sizes[1] / NIMG;             // 1024*1024

    // zero accumulators every call (deterministic across graph replays)
    const int wsFloats = NIMG * WS_STRIDE;
    dl_zero_ws<<<(wsFloats + 255) / 256, 256, 0, stream>>>(ws, wsFloats);

    // pass 1: bf16 wmma segment sums. 256 threads = 8 waves; 4096 px per block
    // (512 px per wave -> 16 wmma iterations of 32 px).
    const int pxPerBlock = 4096;
    const int blocksPerImage = hw / pxPerBlock;
    dl_pass1_wmma<<<dim3(NIMG * blocksPerImage), dim3(256), 0, stream>>>(
        emb, mask, ws, blocksPerImage, pxPerBlock, hw);

    // means (tiny)
    dl_means<<<dim3(NIMG), dim3(96), 0, stream>>>(ws);

    // pass 2: distances + reg
    dl_pass2<<<dim3(NIMG * blocksPerImage / 2), dim3(256), 0, stream>>>(
        emb, mask, ws, blocksPerImage / 2, pxPerBlock * 2, hw);

    // finalize scalar
    dl_finalize<<<dim3(1), dim3(32), 0, stream>>>(ws, out);
}
